// DirectVoxGO_38757784879465
// MI455X (gfx1250) — matrix operand order repaired and checked by reading, not compiled, not run on hardware
//
#include <hip/hip_runtime.h>
#include <math.h>

typedef _Float16 v8h  __attribute__((ext_vector_type(8)));
typedef _Float16 v16h __attribute__((ext_vector_type(16)));
typedef float    v8f  __attribute__((ext_vector_type(8)));

#define GW 100
#define GCH 1000000           // 100*100*100
#define R_RAYS 4096
#define P_PTS 128
#define ACT_SHIFT_F (-13.8155095579643f)  // log(1/(1-1e-6)-1)

// f16 weight-fragment arena layout (element = _Float16):
//  w1 frags: 8 ntiles * 2 ksteps = 16 frags @ 512 halves  -> [0, 8192)
//  w2 frags: 8 ntiles * 4 ksteps = 32 frags @ 512 halves  -> [8192, 24576)
//  w3 frags: 1 ntile  * 4 ksteps =  4 frags @ 512 halves  -> [24576, 26624)
#define W1_OFF 0
#define W2_OFF 8192
#define W3_OFF 24576
#define WF_TOTAL 26624

// B-fragment element mapping (V_WMMA_F32_16X16X32_F16, 16-bit B 32x16):
//   lane L, half j :  n = ntile*16 + (L & 15),  k = kstep*32 + (L>>4)*8 + (j&7) + (j>>3)*16
// Stored frag-major: wf[frag*512 + lane*16 + j]
__global__ void dvgo_pack_weights(const float* __restrict__ w1,
                                  const float* __restrict__ w2,
                                  const float* __restrict__ w3,
                                  _Float16* __restrict__ wf) {
    int tid = blockIdx.x * blockDim.x + threadIdx.x;
    if (tid >= WF_TOTAL) return;
    float val;
    if (tid < W2_OFF) {                       // layer 1: K=39 (pad 64), N=128
        int f = tid >> 9, r = tid & 511;
        int lane = r >> 4, j = r & 15;
        int nt = f >> 1, ks = f & 1;
        int n = nt * 16 + (lane & 15);
        int k = ks * 32 + ((lane >> 4) << 3) + (j & 7) + ((j >> 3) << 4);
        val = (k < 39) ? w1[k * 128 + n] : 0.0f;
    } else if (tid < W3_OFF) {                // layer 2: K=128, N=128
        int e = tid - W2_OFF;
        int f = e >> 9, r = e & 511;
        int lane = r >> 4, j = r & 15;
        int nt = f >> 2, ks = f & 3;
        int n = nt * 16 + (lane & 15);
        int k = ks * 32 + ((lane >> 4) << 3) + (j & 7) + ((j >> 3) << 4);
        val = w2[k * 128 + n];
    } else {                                  // layer 3: K=128, N=3 (pad 16)
        int e = tid - W3_OFF;
        int f = e >> 9, r = e & 511;
        int lane = r >> 4, j = r & 15;
        int ks = f;
        int n = lane & 15;
        int k = ks * 32 + ((lane >> 4) << 3) + (j & 7) + ((j >> 3) << 4);
        val = (n < 3) ? w3[k * 3 + n] : 0.0f;
    }
    wf[tid] = (_Float16)val;
}

__device__ __forceinline__ v16h cat8h(v8h lo, v8h hi) {
    return __builtin_shufflevector(lo, hi, 0, 1, 2, 3, 4, 5, 6, 7,
                                          8, 9, 10, 11, 12, 13, 14, 15);
}

// A-fragment (16-bit A 16x32): lane L -> row m = tile*16 + (L&15);
// halves 0..7 hold k = k0..k0+7, halves 8..15 hold k = k0+16..k0+23,
// with k0 = kstep*32 + (L>>4)*8.
__device__ __forceinline__ v16h load_a(const _Float16* row, int k0) {
    v8h lo = *(const v8h*)(row + k0);
    v8h hi = *(const v8h*)(row + k0 + 16);
    return cat8h(lo, hi);
}

__device__ __forceinline__ v16h load_b(const _Float16* frag, int lane) {
    const v8h* p = (const v8h*)(frag + lane * 16);
    return cat8h(p[0], p[1]);
}

// LDS strides (in halves): X0 row = 72 (K=64 + pad), H row = 136 (K=128 + pad)
#define X0_STR 72
#define H_STR 136
#define SMEM_HALVES (128 * X0_STR + 2 * 128 * H_STR)

__global__ void __launch_bounds__(256)
dvgo_main(const float* __restrict__ origins,
          const float* __restrict__ dirs,
          const float* __restrict__ lengths,
          const float* __restrict__ gd,
          const float* __restrict__ gc,
          const float* __restrict__ b1,
          const float* __restrict__ b2,
          const float* __restrict__ b3,
          const _Float16* __restrict__ wf,
          float* __restrict__ out) {
    extern __shared__ _Float16 smem[];
    _Float16* X0 = smem;                       // [128][72]  layer-0 features
    _Float16* H1 = smem + 128 * X0_STR;        // [128][136] hidden 1
    _Float16* H2 = H1 + 128 * H_STR;           // [128][136] hidden 2

    const int ray = blockIdx.x;
    const int tid = threadIdx.x;               // 0..255
    const int lane = tid & 31;
    const int wv = tid >> 5;                   // 0..7 : one 16-pt tile per wave

    const float dx = dirs[ray * 3 + 0];
    const float dy = dirs[ray * 3 + 1];
    const float dz = dirs[ray * 3 + 2];

    // ---------------- Phase 1: sampling, alpha, feature rows ----------------
    // Two threads per point: half 0 -> density + color ch 0..5 (+alpha),
    //                        half 1 -> color ch 6..11 + view-dir PE + pad.
    {
        const int p = tid & 127;
        const int half = tid >> 7;
        const float t = lengths[ray * P_PTS + p];
        const float x = origins[ray * 3 + 0] + dx * t;
        const float y = origins[ray * 3 + 1] + dy * t;
        const float z = origins[ray * 3 + 2] + dz * t;

        // align_corners=True pixel coords for [-1,1] world (norm == xyz)
        const float px = (x + 1.0f) * 49.5f;
        const float py = (y + 1.0f) * 49.5f;
        const float pz = (z + 1.0f) * 49.5f;
        const float fx0 = floorf(px), fy0 = floorf(py), fz0 = floorf(pz);
        const float fx = px - fx0, fy = py - fy0, fz = pz - fz0;
        const int ix0 = (int)fx0, iy0 = (int)fy0, iz0 = (int)fz0;

        int offs[8];
        float wts[8];
#pragma unroll
        for (int corner = 0; corner < 8; ++corner) {
            const int cx = corner & 1, cy = (corner >> 1) & 1, cz = (corner >> 2) & 1;
            const int ix = ix0 + cx, iy = iy0 + cy, iz = iz0 + cz;
            const bool valid = (ix >= 0) & (ix < GW) & (iy >= 0) & (iy < GW) &
                               (iz >= 0) & (iz < GW);
            const int ixc = min(max(ix, 0), GW - 1);
            const int iyc = min(max(iy, 0), GW - 1);
            const int izc = min(max(iz, 0), GW - 1);
            float w = (cx ? fx : 1.0f - fx) * (cy ? fy : 1.0f - fy) *
                      (cz ? fz : 1.0f - fz);
            wts[corner] = valid ? w : 0.0f;
            offs[corner] = (izc * GW + iyc) * GW + ixc;
        }

        _Float16* xr = X0 + p * X0_STR;
        if (half == 0) {
            float den = 0.0f;
            float col[6];
#pragma unroll
            for (int c = 0; c < 6; ++c) col[c] = 0.0f;
#pragma unroll
            for (int corner = 0; corner < 8; ++corner) {
                const float w = wts[corner];
                const int off = offs[corner];
                den += gd[off] * w;
#pragma unroll
                for (int c = 0; c < 6; ++c) col[c] += gc[c * GCH + off] * w;
            }
            // alpha = 1-(1+exp(den+shift))^(-|d|) = 1-exp(-|d|*softplus(den+shift))
            const float interval = sqrtf(dx * dx + dy * dy + dz * dz);
            float sp = den + ACT_SHIFT_F;
            sp = (sp > 20.0f) ? sp : log1pf(__expf(sp));
            out[ray * P_PTS + p] = 1.0f - __expf(-interval * sp);
#pragma unroll
            for (int c = 0; c < 6; ++c) xr[c] = (_Float16)col[c];
        } else {
            float col[6];
#pragma unroll
            for (int c = 0; c < 6; ++c) col[c] = 0.0f;
#pragma unroll
            for (int corner = 0; corner < 8; ++corner) {
                const float w = wts[corner];
                const int off = offs[corner];
#pragma unroll
                for (int c = 0; c < 6; ++c) col[c] += gc[(c + 6) * GCH + off] * w;
            }
#pragma unroll
            for (int c = 0; c < 6; ++c) xr[6 + c] = (_Float16)col[c];
            xr[12] = (_Float16)dx;
            xr[13] = (_Float16)dy;
            xr[14] = (_Float16)dz;
#pragma unroll
            for (int a = 0; a < 3; ++a) {
                const float d = (a == 0) ? dx : ((a == 1) ? dy : dz);
#pragma unroll
                for (int f = 0; f < 4; ++f) {
                    const float e = d * (float)(1 << f);
                    xr[15 + a * 4 + f] = (_Float16)__sinf(e);
                    xr[27 + a * 4 + f] = (_Float16)__cosf(e);
                }
            }
#pragma unroll
            for (int k = 39; k < 64; ++k) xr[k] = (_Float16)0.0f;
        }
    }
    __syncthreads();

    // ---------------- Phase 2: MLP via v_wmma_f32_16x16x32_f16 ----------------
    // One 16-point tile per wave; A-fragments hoisted out of the ntile loops.
    const _Float16* w1f = wf + W1_OFF;
    const _Float16* w2f = wf + W2_OFF;
    const _Float16* w3f = wf + W3_OFF;
    const int nloc = lane & 15;
    const int g = lane >> 4;
    const int m0 = wv * 16;

    // ---- layer 1: [16x64(f16)] x [64x128] + b1, ReLU ----
    {
        const _Float16* aX = X0 + (m0 + nloc) * X0_STR;
        const v16h a0 = load_a(aX, g * 8);
        const v16h a1 = load_a(aX, 32 + g * 8);
#pragma unroll
        for (int nt = 0; nt < 8; ++nt) {
            const float bias = b1[nt * 16 + nloc];
            v8f acc;
#pragma unroll
            for (int i = 0; i < 8; ++i) acc[i] = bias;
            acc = __builtin_amdgcn_wmma_f32_16x16x32_f16(
                false, a0, false, load_b(w1f + (nt * 2 + 0) * 512, lane),
                (short)0, acc, false, false);
            acc = __builtin_amdgcn_wmma_f32_16x16x32_f16(
                false, a1, false, load_b(w1f + (nt * 2 + 1) * 512, lane),
                (short)0, acc, false, false);
#pragma unroll
            for (int i = 0; i < 8; ++i) {
                H1[(m0 + g * 8 + i) * H_STR + nt * 16 + nloc] =
                    (_Float16)fmaxf(acc[i], 0.0f);
            }
        }
    }

    // ---- layer 2: [16x128] x [128x128] + b2, ReLU ----
    {
        const _Float16* aH1 = H1 + (m0 + nloc) * H_STR;
        v16h a[4];
#pragma unroll
        for (int ks = 0; ks < 4; ++ks) a[ks] = load_a(aH1, ks * 32 + g * 8);
#pragma unroll
        for (int nt = 0; nt < 8; ++nt) {
            const float bias = b2[nt * 16 + nloc];
            v8f acc;
#pragma unroll
            for (int i = 0; i < 8; ++i) acc[i] = bias;
#pragma unroll
            for (int ks = 0; ks < 4; ++ks) {
                acc = __builtin_amdgcn_wmma_f32_16x16x32_f16(
                    false, a[ks], false, load_b(w2f + (nt * 4 + ks) * 512, lane),
                    (short)0, acc, false, false);
            }
#pragma unroll
            for (int i = 0; i < 8; ++i) {
                H2[(m0 + g * 8 + i) * H_STR + nt * 16 + nloc] =
                    (_Float16)fmaxf(acc[i], 0.0f);
            }
        }
    }

    // ---- layer 3: [16x128] x [128x16(pad of 3)] + b3, sigmoid ----
    {
        const _Float16* aH2 = H2 + (m0 + nloc) * H_STR;
        v16h a[4];
#pragma unroll
        for (int ks = 0; ks < 4; ++ks) a[ks] = load_a(aH2, ks * 32 + g * 8);
        const float bias = (nloc < 3) ? b3[nloc] : 0.0f;
        v8f acc;
#pragma unroll
        for (int i = 0; i < 8; ++i) acc[i] = bias;
#pragma unroll
        for (int ks = 0; ks < 4; ++ks) {
            acc = __builtin_amdgcn_wmma_f32_16x16x32_f16(
                false, a[ks], false, load_b(w3f + ks * 512, lane),
                (short)0, acc, false, false);
        }
        if (nloc < 3) {
#pragma unroll
            for (int i = 0; i < 8; ++i) {
                const float v = 1.0f / (1.0f + __expf(-acc[i]));
                const int p = m0 + g * 8 + i;
                out[R_RAYS * P_PTS + (ray * P_PTS + p) * 3 + nloc] = v;
            }
        }
    }
}

extern "C" void kernel_launch(void* const* d_in, const int* in_sizes, int n_in,
                              void* d_out, int out_size, void* d_ws, size_t ws_size,
                              hipStream_t stream) {
    (void)in_sizes; (void)n_in; (void)out_size; (void)ws_size;
    const float* origins = (const float*)d_in[0];
    const float* dirs    = (const float*)d_in[1];
    const float* lengths = (const float*)d_in[2];
    const float* gd      = (const float*)d_in[3];
    const float* gc      = (const float*)d_in[4];
    const float* w1      = (const float*)d_in[5];
    const float* b1      = (const float*)d_in[6];
    const float* w2      = (const float*)d_in[7];
    const float* b2      = (const float*)d_in[8];
    const float* w3      = (const float*)d_in[9];
    const float* b3      = (const float*)d_in[10];
    float* out = (float*)d_out;
    _Float16* wf = (_Float16*)d_ws;

    dvgo_pack_weights<<<(WF_TOTAL + 255) / 256, 256, 0, stream>>>(w1, w2, w3, wf);
    dvgo_main<<<R_RAYS, 256, SMEM_HALVES * sizeof(_Float16), stream>>>(
        origins, dirs, lengths, gd, gc, b1, b2, b3, wf, out);
}
